// DisentangledLSTMAutoEncoder_17051020165693
// MI455X (gfx1250) — compile-verified, matrix operand order checked by reading, not run on hardware
//
#include <hip/hip_runtime.h>
#include <hip/hip_bf16.h>

// ---------------------------------------------------------------------------
// Disentangled LSTM autoencoder, fused per-layer persistent scan kernels.
// B=256, S=512, IN=64, H=32, REP=64, SH=SP=32.
//
// Input order (setup_inputs dict, depth-first):
//  0:x 1-4:enc1{Wih,Whh,bih,bhh} 5-8:enc2{...} 9-12:dec1{...} 13-16:dec2{...}
//  17:W_share 18:b_share 19:W_spec 20:b_spec
// Output: x_recon (B*S*64) | z_share (B*32) | z_spec (B*32), fp32.
// ---------------------------------------------------------------------------

typedef __attribute__((ext_vector_type(16))) __bf16 v16bf;
typedef __attribute__((ext_vector_type(8)))  float  v8f;

#define B_SZ 256
#define S_SZ 512

// A/B fragment for v_wmma_f32_16x16x32_bf16.
// Per-lane layout (ISA 7.12.2, 16-bit A 16x32): lanes 0-15 row m=lane, K runs
// [0,8) and [16,24); lanes 16-31 same m=lane-16, K runs [8,16) and [24,32).
// Element e<8 -> K=kb+e ; e>=8 -> K=kb+16+(e-8).
__device__ __forceinline__ v16bf load_frag_f32(const float* __restrict__ p, int kb) {
  v16bf v;
#pragma unroll
  for (int i = 0; i < 8; ++i) {
    v[i]     = (__bf16)p[kb + i];
    v[i + 8] = (__bf16)p[kb + 16 + i];
  }
  return v;
}

__device__ __forceinline__ v16bf load_frag_bf16(const __bf16* __restrict__ p, int kb) {
  v16bf v;
#pragma unroll
  for (int i = 0; i < 8; ++i) {
    v[i]     = p[kb + i];
    v[i + 8] = p[kb + 16 + i];
  }
  return v;
}

// Branchless activations on the hardware transcendental pipe (v_exp_f32 /
// v_rcp_f32). Forms are chosen to be NaN-free and exact at the saturated
// limits, so no EXEC-divergent range splits appear on the scan's critical
// dependence chain.
#define LOG2E_F  1.442695041f
#define LOG2E2_F 2.885390082f

__device__ __forceinline__ float fast_sigmoid(float x) {
  // x->+inf: exp2(-inf)=0 -> rcp(1)=1 ; x->-inf: exp2(+inf)=inf -> rcp(inf)=0
  return __builtin_amdgcn_rcpf(1.0f + __builtin_amdgcn_exp2f(-x * LOG2E_F));
}

__device__ __forceinline__ float fast_tanh(float x) {
  // tanh(x) = 1 - 2/(e^{2x}+1):
  // x->+inf: exp2 overflows to inf -> rcp(inf)=0 -> 1
  // x->-inf: exp2 underflows to 0  -> 1 - 2*rcp(1) = -1
  return 1.0f - 2.0f * __builtin_amdgcn_rcpf(__builtin_amdgcn_exp2f(x * LOG2E2_F) + 1.0f);
}

// One workgroup = one 16-row batch tile for the WHOLE sequence.
// Wave w owns gate columns [16w, 16w+16). Weight B-fragments pinned in VGPRs.
// h: bf16 in LDS (WMMA A operand). c: fp32 in registers. gates: fp32 in LDS.
template <int HID, int KIN>
__global__ __launch_bounds__(8 * HID)
void lstm_scan_kernel(const float* __restrict__ in, long inRowStr, long inSStr,
                      const float* __restrict__ Wih, const float* __restrict__ Whh,
                      const float* __restrict__ bih, const float* __restrict__ bhh,
                      float* __restrict__ out, long outRowStr, long outSStr,
                      int seqLen, int storeAll)
{
  constexpr int NG = 4 * HID;    // gate columns
  constexpr int NW = NG / 16;    // waves per block
  constexpr int NT = NW * 32;    // threads per block
  constexpr int KI = KIN / 32;   // K-steps over input dim
  constexpr int KH = HID / 32;   // K-steps over hidden dim
  static_assert(16 * HID == 2 * NT, "two cell elements per thread");

  __shared__ __bf16 hbuf[16][HID];
  __shared__ float  gbuf[16][NG];

  const int tid  = threadIdx.x;
  const int lane = tid & 31;
  const int wave = tid >> 5;
  const int n    = lane & 15;
  const int hi   = (lane >> 4) & 1;
  const int kb   = hi ? 8 : 0;   // K-run base for A/B fragments
  const int mb   = hi ? 8 : 0;   // row base for C/D tile
  const int col  = wave * 16 + n;    // this lane's gate column
  const int m    = n;                // this lane's A row within the tile
  const int bt   = blockIdx.x;       // batch tile (16 rows)

  // --- Pin weight (B) fragments in registers: B[k][col] = W[col][k]. ---
  v16bf Bi[KI];
#pragma unroll
  for (int kk = 0; kk < KI; ++kk)
    Bi[kk] = load_frag_f32(Wih + (long)col * KIN + kk * 32, kb);
  v16bf Bh[KH];
#pragma unroll
  for (int kk = 0; kk < KH; ++kk)
    Bh[kk] = load_frag_f32(Whh + (long)col * HID + kk * 32, kb);

  const float bias = bih[col] + bhh[col];

  // Per-thread cell state: elements e0, e1 of the 16 x HID (m,u) grid.
  const int e0 = tid, e1 = tid + NT;
  const int m0 = e0 / HID, u0 = e0 % HID;
  const int m1 = e1 / HID, u1 = e1 % HID;
  float c0 = 0.f, c1 = 0.f;
  hbuf[m0][u0] = (__bf16)0.f;
  hbuf[m1][u1] = (__bf16)0.f;
  __syncthreads();

  const float* aBase = in + (long)(bt * 16 + m) * inRowStr;

  for (int s = 0; s < seqLen; ++s) {
    const float* arow = aBase + (long)s * inSStr;
    if (s + 1 < seqLen) __builtin_prefetch(arow + inSStr, 0, 1);

    v8f acc;
#pragma unroll
    for (int r = 0; r < 8; ++r) acc[r] = bias;

    // gates_tile = x_t @ Wih^T  (+ bias, in acc)
#pragma unroll
    for (int kk = 0; kk < KI; ++kk) {
      v16bf a = load_frag_f32(arow + kk * 32, kb);
      acc = __builtin_amdgcn_wmma_f32_16x16x32_bf16(
          false, a, false, Bi[kk], (short)0, acc, false, false);
    }
    //            + h_{t-1} @ Whh^T
#pragma unroll
    for (int kk = 0; kk < KH; ++kk) {
      v16bf a = load_frag_bf16(&hbuf[m][kk * 32], kb);
      acc = __builtin_amdgcn_wmma_f32_16x16x32_bf16(
          false, a, false, Bh[kk], (short)0, acc, false, false);
    }

    // C/D tile scatter: VGPR r -> row mb+r, col = this lane's column.
#pragma unroll
    for (int r = 0; r < 8; ++r) gbuf[mb + r][col] = acc[r];
    __syncthreads();

    // Elementwise cell update (gate order i,f,g,o along the 4H axis).
    {
      float ig = fast_sigmoid(gbuf[m0][u0]);
      float fg = fast_sigmoid(gbuf[m0][HID + u0]);
      float gg = fast_tanh(gbuf[m0][2 * HID + u0]);
      float og = fast_sigmoid(gbuf[m0][3 * HID + u0]);
      c0 = fg * c0 + ig * gg;
      float h = og * fast_tanh(c0);
      hbuf[m0][u0] = (__bf16)h;
      if (storeAll || s == seqLen - 1)
        out[(long)(bt * 16 + m0) * outRowStr + (long)s * outSStr + u0] = h;
    }
    {
      float ig = fast_sigmoid(gbuf[m1][u1]);
      float fg = fast_sigmoid(gbuf[m1][HID + u1]);
      float gg = fast_tanh(gbuf[m1][2 * HID + u1]);
      float og = fast_sigmoid(gbuf[m1][3 * HID + u1]);
      c1 = fg * c1 + ig * gg;
      float h = og * fast_tanh(c1);
      hbuf[m1][u1] = (__bf16)h;
      if (storeAll || s == seqLen - 1)
        out[(long)(bt * 16 + m1) * outRowStr + (long)s * outSStr + u1] = h;
    }
    __syncthreads();
  }
}

// z_share / z_spec heads: exact fp32 dot products (tiny: 2 x 256x32, K=64).
__global__ void head_kernel(const float* __restrict__ hlast,
                            const float* __restrict__ Wsh, const float* __restrict__ bsh,
                            const float* __restrict__ Wsp, const float* __restrict__ bsp,
                            float* __restrict__ zsh, float* __restrict__ zsp)
{
  int t = blockIdx.x * blockDim.x + threadIdx.x;
  if (t >= B_SZ * 64) return;
  int b = t >> 6, j = t & 63;
  const float* h = hlast + b * 64;
  const float* w; const float* bb; float* z; int jj;
  if (j < 32) { jj = j;      w = Wsh + (long)jj * 64; bb = bsh; z = zsh; }
  else        { jj = j - 32; w = Wsp + (long)jj * 64; bb = bsp; z = zsp; }
  float acc = bb[jj];
#pragma unroll
  for (int k = 0; k < 64; ++k) acc += h[k] * w[k];
  z[(long)b * 32 + jj] = acc;
}

extern "C" void kernel_launch(void* const* d_in, const int* in_sizes, int n_in,
                              void* d_out, int out_size, void* d_ws, size_t ws_size,
                              hipStream_t stream)
{
  (void)in_sizes; (void)n_in; (void)out_size; (void)ws_size;

  const float* x     = (const float*)d_in[0];
  const float* Wih1  = (const float*)d_in[1];
  const float* Whh1  = (const float*)d_in[2];
  const float* bih1  = (const float*)d_in[3];
  const float* bhh1  = (const float*)d_in[4];
  const float* Wih2  = (const float*)d_in[5];
  const float* Whh2  = (const float*)d_in[6];
  const float* bih2  = (const float*)d_in[7];
  const float* bhh2  = (const float*)d_in[8];
  const float* Wihd1 = (const float*)d_in[9];
  const float* Whhd1 = (const float*)d_in[10];
  const float* bihd1 = (const float*)d_in[11];
  const float* bhhd1 = (const float*)d_in[12];
  const float* Wihd2 = (const float*)d_in[13];
  const float* Whhd2 = (const float*)d_in[14];
  const float* bihd2 = (const float*)d_in[15];
  const float* bhhd2 = (const float*)d_in[16];
  const float* Wsh   = (const float*)d_in[17];
  const float* bsh   = (const float*)d_in[18];
  const float* Wsp   = (const float*)d_in[19];
  const float* bsp   = (const float*)d_in[20];

  float* out    = (float*)d_out;
  float* xrecon = out;                              // B*S*64
  float* zsh    = out + (long)B_SZ * S_SZ * 64;     // B*32
  float* zsp    = zsh + (long)B_SZ * 32;            // B*32

  float* ws    = (float*)d_ws;
  float* hs1   = ws;                                // B*S*32  (enc1 output)
  float* hlast = hs1 + (long)B_SZ * S_SZ * 32;      // B*64    (enc2 last h)
  float* hsd1  = hlast + (long)B_SZ * 64;           // B*S*32  (dec1 output)

  const dim3 grid(B_SZ / 16);

  // enc1: (B,S,64) -> hs1 (B,S,32)
  lstm_scan_kernel<32, 64><<<grid, 256, 0, stream>>>(
      x, (long)S_SZ * 64, 64, Wih1, Whh1, bih1, bhh1,
      hs1, (long)S_SZ * 32, 32, S_SZ, 1);

  // enc2: hs1 -> h_last (B,64), only final step stored
  lstm_scan_kernel<64, 32><<<grid, 512, 0, stream>>>(
      hs1, (long)S_SZ * 32, 32, Wih2, Whh2, bih2, bhh2,
      hlast, 64, 0, S_SZ, 0);

  // heads: z_share, z_spec -> d_out tail
  head_kernel<<<(B_SZ * 64 + 255) / 256, 256, 0, stream>>>(
      hlast, Wsh, bsh, Wsp, bsp, zsh, zsp);

  // dec1: time-constant input z_spec (sStride=0) -> hsd1 (B,S,32)
  lstm_scan_kernel<32, 32><<<grid, 256, 0, stream>>>(
      zsp, 32, 0, Wihd1, Whhd1, bihd1, bhhd1,
      hsd1, (long)S_SZ * 32, 32, S_SZ, 1);

  // dec2: hsd1 -> x_recon (B,S,64) directly into d_out
  lstm_scan_kernel<64, 32><<<grid, 512, 0, stream>>>(
      hsd1, (long)S_SZ * 32, 32, Wihd2, Whhd2, bihd2, bhhd2,
      xrecon, (long)S_SZ * 64, 64, S_SZ, 1);
}
